// TransportOperator_58884001628184
// MI455X (gfx1250) — matrix operand order repaired and checked by reading, not compiled, not run on hardware
//
#include <hip/hip_runtime.h>
#include <math.h>

// ---------------------------------------------------------------------------
// TransportOperator forward, MI455X (gfx1250, wave32, WMMA).
//
//   W[m,d]     = sum_e Psi[m,d,e] * g[m,e]
//   gates[b,m] = relu(2*sigmoid(z[b,:]·W[m,:]) - 1)
//   dz[b,e]    = sum_{m,d} (gates[b,m]*z[b,d]) * Psi[m,d,e]   (single K=8192 GEMM)
// ---------------------------------------------------------------------------

typedef __attribute__((ext_vector_type(16))) _Float16 v16h;
typedef __attribute__((ext_vector_type(8)))  _Float16 v8h;
typedef __attribute__((ext_vector_type(4)))  _Float16 v4h;
typedef __attribute__((ext_vector_type(8)))  float    v8f;
typedef __attribute__((ext_vector_type(2)))  float    v2f;

#define NB 16384
#define ND 512
#define NM 16

// ---------------------------------------------------------------- z -> f16
__global__ __launch_bounds__(256) void prep_zh(const float* __restrict__ z,
                                               _Float16* __restrict__ zh) {
    int idx = (blockIdx.x * 256 + threadIdx.x) * 4;   // B*D / (256*4) blocks, exact
    float4 v = *(const float4*)(z + idx);
    v4h h;
    h[0] = (_Float16)v.x; h[1] = (_Float16)v.y;
    h[2] = (_Float16)v.z; h[3] = (_Float16)v.w;
    *(v4h*)(zh + idx) = h;
}

// ------------------------------------------- Psi[m,d,e] -> f16 Psit[m,e,d]
__global__ __launch_bounds__(256) void prep_psit(const float* __restrict__ Psi,
                                                 _Float16* __restrict__ pt) {
    __shared__ _Float16 tile[32][33];
    int m  = blockIdx.z;
    int e0 = blockIdx.x * 32;
    int d0 = blockIdx.y * 32;
    int tx = threadIdx.x;          // 0..31
    int ty = threadIdx.y;          // 0..7
    const float* src = Psi + ((size_t)m * ND) * ND;
#pragma unroll
    for (int j = 0; j < 4; ++j) {
        int d = d0 + ty + 8 * j;
        tile[ty + 8 * j][tx] = (_Float16)src[(size_t)d * ND + (e0 + tx)];
    }
    __syncthreads();
    _Float16* dst = pt + ((size_t)m * ND) * ND;
#pragma unroll
    for (int j = 0; j < 4; ++j) {
        int e = e0 + ty + 8 * j;
        dst[(size_t)e * ND + (d0 + tx)] = tile[tx][ty + 8 * j];
    }
}

// ------------------------------------------------ W[m,d] = Psi[m,d,:]·g[m,:]
__global__ __launch_bounds__(256) void prep_W(const float* __restrict__ Psi,
                                              const float* __restrict__ g,
                                              float* __restrict__ W) {
    int wid  = blockIdx.x * 8 + (threadIdx.x >> 5);   // one wave per output
    int lane = threadIdx.x & 31;
    int m = wid >> 9;          // / 512
    int d = wid & 511;
    const float* pr = Psi + ((size_t)m * ND + d) * ND;
    const float* gr = g + (size_t)m * ND;
    float acc = 0.f;
#pragma unroll
    for (int j = 0; j < 16; ++j) {
        int e = lane + 32 * j;
        acc += pr[e] * gr[e];
    }
#pragma unroll
    for (int s = 16; s >= 1; s >>= 1) acc += __shfl_xor(acc, s, 32);
    if (lane == 0) W[(size_t)m * ND + d] = acc;
}

// -------------------------------------- gates[b,m] = relu(2*sig(z·W) - 1)
__global__ __launch_bounds__(256) void gates_k(const float* __restrict__ z,
                                               const float* __restrict__ W,
                                               float* __restrict__ gates) {
    int b = blockIdx.x * 16 + (threadIdx.x >> 4);
    int m = threadIdx.x & 15;
    const float* zr = z + (size_t)b * ND;
    const float* wr = W + (size_t)m * ND;
    float acc = 0.f;
    for (int d = 0; d < ND; ++d) acc += zr[d] * wr[d];
    float s = 1.f / (1.f + expf(-acc));
    gates[(size_t)b * NM + m] = fmaxf(0.f, 2.f * s - 1.f);
}

// ---------------------------------------------------------------------------
// Main fused GEMM:  dz[128x256 tile per WG] = A(gated z, f16) x Psit (f16)
// 8 waves; each wave owns a 64x64 sub-tile = 4x4 WMMA accumulators.
// A layout (16-bit 16x32): lane=row; lanes<16 K0-15, lanes>=16 K16-31 (same k order as B).
// B layout (16-bit 32x16): lane=col; lanes<16 K0-15, lanes>=16 K16-31.
// ---------------------------------------------------------------------------
__global__ __launch_bounds__(256) void main_f16(const _Float16* __restrict__ zh,
                                                const _Float16* __restrict__ pt,
                                                const float* __restrict__ gates,
                                                float* __restrict__ out) {
    const int lane = threadIdx.x & 31;
    const int w    = threadIdx.x >> 5;
    const int lh   = lane & 15;
    const int hi   = lane >> 4;
    const int wr   = w >> 2;            // 0..1
    const int wc   = w & 3;             // 0..3
    const int rowBase = blockIdx.x * 128 + wr * 64;
    const int colBase = blockIdx.y * 256 + wc * 64;

    v8f acc[4][4];
#pragma unroll
    for (int r = 0; r < 4; ++r)
#pragma unroll
        for (int c = 0; c < 4; ++c) acc[r][c] = (v8f)0.f;

    const _Float16* zrow[4];
#pragma unroll
    for (int r = 0; r < 4; ++r)
        zrow[r] = zh + (size_t)(rowBase + r * 16 + lh) * ND;

    for (int m = 0; m < NM; ++m) {
        _Float16 gh[4];
#pragma unroll
        for (int r = 0; r < 4; ++r)
            gh[r] = (_Float16)gates[(size_t)(rowBase + r * 16 + lh) * NM + m];

        const _Float16* bcol[4];
#pragma unroll
        for (int c = 0; c < 4; ++c)
            bcol[c] = pt + ((size_t)m * ND + (colBase + c * 16 + lh)) * ND + 16 * hi;

        for (int d = 0; d < ND; d += 32) {
            // ---- A fragments: gates[row,m] * zh[row, d..d+31]
            v16h A[4];
#pragma unroll
            for (int r = 0; r < 4; ++r) {
                v8h a0 = *(const v8h*)(zrow[r] + d + 16 * hi);       // K 0-7 / 16-23
                v8h a1 = *(const v8h*)(zrow[r] + d + 16 * hi + 8);   // K 8-15 / 24-31
                v16h a;
#pragma unroll
                for (int i = 0; i < 8; ++i) {
                    a[i]     = a0[i] * gh[r];
                    a[8 + i] = a1[i] * gh[r];
                }
                A[r] = a;
            }
            // ---- B fragments: Psit[m, col, d + 16*hi .. +15] (contiguous)
            v16h Bf[4];
#pragma unroll
            for (int c = 0; c < 4; ++c)
                Bf[c] = *(const v16h*)(bcol[c] + d);
            // ---- 16 WMMAs
#pragma unroll
            for (int r = 0; r < 4; ++r)
#pragma unroll
                for (int c = 0; c < 4; ++c)
                    acc[r][c] = __builtin_amdgcn_wmma_f32_16x16x32_f16(
                        false, A[r], false, Bf[c], (short)0, acc[r][c], false, false);
        }
    }

    // C/D layout: lane=col (mod 16); vgpr i -> row i + 8*hi
#pragma unroll
    for (int r = 0; r < 4; ++r)
#pragma unroll
        for (int c = 0; c < 4; ++c) {
            float* op = out + (size_t)(rowBase + r * 16 + 8 * hi) * ND
                            + colBase + c * 16 + lh;
#pragma unroll
            for (int i = 0; i < 8; ++i) op[(size_t)i * ND] = acc[r][c][i];
        }
}

// ---------------------------------------------------------------------------
// Fallback: f32 WMMA (16x16x4) straight from f32 z / Psi; needs only ~1.1 MB ws.
// 8 waves per WG; WG tile 64x256; wave tile 32x64 (2x4 accumulators).
// ---------------------------------------------------------------------------
__global__ __launch_bounds__(256) void main_f32(const float* __restrict__ z,
                                                const float* __restrict__ Psi,
                                                const float* __restrict__ gates,
                                                float* __restrict__ out) {
    const int lane = threadIdx.x & 31;
    const int w    = threadIdx.x >> 5;
    const int lh   = lane & 15;
    const int hi   = lane >> 4;
    const int wr   = w >> 2;            // 0..1
    const int wc   = w & 3;             // 0..3
    const int rowBase = blockIdx.x * 64 + wr * 32;
    const int colBase = blockIdx.y * 256 + wc * 64;

    v8f acc[2][4];
#pragma unroll
    for (int r = 0; r < 2; ++r)
#pragma unroll
        for (int c = 0; c < 4; ++c) acc[r][c] = (v8f)0.f;

    const float* zrow[2];
#pragma unroll
    for (int r = 0; r < 2; ++r)
        zrow[r] = z + (size_t)(rowBase + r * 16 + lh) * ND;

    for (int m = 0; m < NM; ++m) {
        float gf[2];
#pragma unroll
        for (int r = 0; r < 2; ++r)
            gf[r] = gates[(size_t)(rowBase + r * 16 + lh) * NM + m];
        const float* pm = Psi + (size_t)m * ND * ND;

        for (int d = 0; d < ND; d += 4) {
            v2f A[2];
#pragma unroll
            for (int r = 0; r < 2; ++r) {
                v2f a = *(const v2f*)(zrow[r] + d + 2 * hi);   // K = 2*hi, 2*hi+1
                a.x *= gf[r]; a.y *= gf[r];
                A[r] = a;
            }
            v2f Bf[4];
#pragma unroll
            for (int c = 0; c < 4; ++c) {
                const float* p = pm + (size_t)(d + 2 * hi) * ND + colBase + c * 16 + lh;
                v2f b; b.x = p[0]; b.y = p[ND];
                Bf[c] = b;
            }
#pragma unroll
            for (int r = 0; r < 2; ++r)
#pragma unroll
                for (int c = 0; c < 4; ++c)
                    acc[r][c] = __builtin_amdgcn_wmma_f32_16x16x4_f32(
                        false, A[r], false, Bf[c], (short)0, acc[r][c], false, false);
        }
    }

#pragma unroll
    for (int r = 0; r < 2; ++r)
#pragma unroll
        for (int c = 0; c < 4; ++c) {
            float* op = out + (size_t)(rowBase + r * 16 + 8 * hi) * ND
                            + colBase + c * 16 + lh;
#pragma unroll
            for (int i = 0; i < 8; ++i) op[(size_t)i * ND] = acc[r][c][i];
        }
}

// ---------------------------------------------------------------------------
extern "C" void kernel_launch(void* const* d_in, const int* in_sizes, int n_in,
                              void* d_out, int out_size, void* d_ws, size_t ws_size,
                              hipStream_t stream) {
    (void)in_sizes; (void)n_in; (void)out_size;
    const float* z   = (const float*)d_in[1];   // [B,D]
    const float* Psi = (const float*)d_in[2];   // [M,D,D]
    const float* g   = (const float*)d_in[3];   // [M,D]
    float* out = (float*)d_out;                 // [B,D]

    const size_t zh_b = (size_t)NB * ND * 2;        // 16 MB
    const size_t pt_b = (size_t)NM * ND * ND * 2;   // 8 MB
    const size_t w_b  = (size_t)NM * ND * 4;        // 32 KB
    const size_t gt_b = (size_t)NB * NM * 4;        // 1 MB
    char* ws = (char*)d_ws;

    if (ws_size >= zh_b + pt_b + w_b + gt_b) {
        _Float16* zh    = (_Float16*)ws;
        _Float16* pt    = (_Float16*)(ws + zh_b);
        float*    W     = (float*)(ws + zh_b + pt_b);
        float*    gates = (float*)(ws + zh_b + pt_b + w_b);

        prep_zh  <<<(NB * ND) / (256 * 4), 256, 0, stream>>>(z, zh);
        prep_psit<<<dim3(ND / 32, ND / 32, NM), dim3(32, 8), 0, stream>>>(Psi, pt);
        prep_W   <<<(NM * ND) / 8 / 32, 256, 0, stream>>>(Psi, g, W);
        gates_k  <<<NB / 16, 256, 0, stream>>>(z, W, gates);
        main_f16 <<<dim3(NB / 128, ND / 256), 256, 0, stream>>>(zh, pt, gates, out);
    } else {
        float* W     = (float*)ws;
        float* gates = (float*)(ws + w_b);
        prep_W <<<(NM * ND) / 8 / 32, 256, 0, stream>>>(Psi, g, W);
        gates_k<<<NB / 16, 256, 0, stream>>>(z, W, gates);
        main_f32<<<dim3(NB / 64, ND / 256), 256, 0, stream>>>(z, Psi, gates, out);
    }
}
